// ParametricSynth_44178033607063
// MI455X (gfx1250) — matrix operand order repaired and checked by reading, not compiled, not run on hardware
//
#include <hip/hip_runtime.h>
#include <hip/hip_bf16.h>
#include <math.h>

// ---------------------------------------------------------------------------
// Griffin-Lim parametric synth on MI455X (gfx1250).
// All FFTs are expressed as f16 WMMA GEMMs (v_wmma_f32_16x16x32_f16) against
// precomputed DFT matrices (window / 1/N / irfft semantics folded in).
// ~585 GFLOP total, working set ~170MB (L2-resident on 192MB L2).
// ---------------------------------------------------------------------------

typedef __attribute__((ext_vector_type(16))) _Float16 v16h;
typedef __attribute__((ext_vector_type(8)))  _Float16 v8h;
typedef __attribute__((ext_vector_type(8)))  float    v8f;

#define B_SZ        32
#define N_MELS      128
#define N_FRAMES    256
#define FREQ_BINS   513
#define KHALF       528          // 513 padded to 33*16
#define KSPEC       1056         // 2*KHALF (Re|Im), multiple of 32
#define NSPEC_PAD   1088         // KSPEC padded to 17*64 for B-row loads
#define NFFT        1024
#define HOP         256
#define OUT_LEN     65280        // HOP*(N_FRAMES-1)
#define FULL_LEN    66304        // NFFT + HOP*(N_FRAMES-1)
#define PADL        512
#define NUM_SAMPLES 65536
#define MROWS       8192         // B_SZ*N_FRAMES
#define N_ITER      16
#define MOMC        0.497487437185929648f   // 0.99/1.99
#define TWO_PI      6.28318530717958647692f

static __host__ __device__ inline int cdiv(int a, int b) { return (a + b - 1) / b; }

// ------------------------- WMMA GEMM: C = A * B^T' -------------------------
// A: [M x K] f16 row-major (lda). Bt: [>=Nvalid x K] f16 row-major (ldb),
// i.e. B stored transposed so fragment loads are contiguous. C: f32 [M x ldc].
// One wave computes a 64x64 tile: 4x4 fragments of 16x16, K-steps of 32.
__global__ void __launch_bounds__(256)
gemm_wmma(const _Float16* __restrict__ A, const _Float16* __restrict__ Bt,
          float* __restrict__ C, int Nvalid, int K,
          int lda, int ldb, int ldc, int Mt, int Nt) {
  int wid = blockIdx.x * (blockDim.x >> 5) + (threadIdx.x >> 5);
  if (wid >= Mt * Nt) return;
  int tm = wid % Mt, tn = wid / Mt;
  int m0 = tm * 64, n0 = tn * 64;
  int lane = threadIdx.x & 31;
  int l15 = lane & 15, g = lane >> 4;

  v8f acc[4][4] = {};
  const _Float16* Arow[4];
  const _Float16* Brow[4];
#pragma unroll
  for (int i = 0; i < 4; ++i)
    Arow[i] = A + (size_t)(m0 + 16 * i + l15) * lda + 8 * g;
#pragma unroll
  for (int j = 0; j < 4; ++j)
    Brow[j] = Bt + (size_t)(n0 + 16 * j + l15) * ldb + 8 * g;

  union HV { v16h v; v8h h[2]; };

  for (int kk = 0; kk < K; kk += 32) {
    v16h aF[4], bF[4];
#pragma unroll
    for (int i = 0; i < 4; ++i) {
      HV u;
      u.h[0] = *(const v8h*)(Arow[i] + kk);        // K = kk+8g .. +7
      u.h[1] = *(const v8h*)(Arow[i] + kk + 16);   // K = kk+16+8g .. +7
      aF[i] = u.v;
    }
#pragma unroll
    for (int j = 0; j < 4; ++j) {
      HV u;
      u.h[0] = *(const v8h*)(Brow[j] + kk);        // K = kk+8g .. +7
      u.h[1] = *(const v8h*)(Brow[j] + kk + 16);   // K = kk+16+8g .. +7
      bF[j] = u.v;
    }
#pragma unroll
    for (int i = 0; i < 4; ++i)
#pragma unroll
      for (int j = 0; j < 4; ++j)
        acc[i][j] = __builtin_amdgcn_wmma_f32_16x16x32_f16(
            false, aF[i], false, bF[j], (short)0, acc[i][j], false, false);
  }

#pragma unroll
  for (int i = 0; i < 4; ++i)
#pragma unroll
    for (int j = 0; j < 4; ++j) {
      int n = n0 + 16 * j + l15;
      if (n0 + 16 * j >= Nvalid) continue;
      float* cp = C + (size_t)(m0 + 16 * i + 8 * g) * ldc + n;
#pragma unroll
      for (int r = 0; r < 8; ++r) cp[(size_t)r * ldc] = acc[i][j][r];
    }
}

// ------------------------- setup: window + DFT matrices --------------------
__global__ void win_kernel(float* win) {
  int n = blockIdx.x * blockDim.x + threadIdx.x;
  if (n < NFFT) win[n] = 0.5f * (1.f - cosf((float)n * (TWO_PI / 1024.f)));
}

// iSTFT matrix, stored transposed: DmatT[n(time) 0..1023][r(spec) 0..1055]
__global__ void build_dmatT(const float* __restrict__ win, _Float16* __restrict__ D) {
  int idx = blockIdx.x * blockDim.x + threadIdx.x;
  if (idx >= NFFT * KSPEC) return;
  int n = idx / KSPEC, r = idx % KSPEC;
  int k = (r < KHALF) ? r : r - KHALF;
  float v = 0.f;
  if (k <= 512) {
    float w = win[n] * (1.0f / 1024.0f);
    if (r < KHALF) {                         // Re coefficients
      if (k == 0)        v = w;
      else if (k == 512) v = (n & 1) ? -w : w;
      else {
        int ph = (k * n) & 1023;
        v = 2.f * cosf((float)ph * (TWO_PI / 1024.f)) * w;
      }
    } else {                                 // Im coefficients (0 for k=0,512)
      if (k >= 1 && k <= 511) {
        int ph = (k * n) & 1023;
        v = -2.f * sinf((float)ph * (TWO_PI / 1024.f)) * w;
      }
    }
  }
  D[idx] = (_Float16)v;
}

// STFT matrix, stored transposed: CmatT[r(spec) 0..1087][n(time) 0..1023]
__global__ void build_cmatT(const float* __restrict__ win, _Float16* __restrict__ Cm) {
  int idx = blockIdx.x * blockDim.x + threadIdx.x;
  if (idx >= NSPEC_PAD * NFFT) return;
  int r = idx / NFFT, j = idx % NFFT;
  float v = 0.f;
  if (r < KSPEC) {
    int k = (r < KHALF) ? r : r - KHALF;
    if (k <= 512) {
      int ph = (k * j) & 1023;
      float ang = (float)ph * (TWO_PI / 1024.f);
      float w = win[j];
      v = (r < KHALF) ? cosf(ang) * w : -sinf(ang) * w;
    }
  }
  Cm[idx] = (_Float16)v;
}

// ------------------------- spectrogram generation --------------------------
__global__ void spec_kernel(const float* __restrict__ params,
                            const float* __restrict__ noise,
                            float* __restrict__ specp) {
  int idx = blockIdx.x * blockDim.x + threadIdx.x;
  if (idx >= B_SZ * N_MELS * N_FRAMES) return;
  int t = idx & 255, mel = (idx >> 8) & 127, b = idx >> 15;
  const float lo[15] = {0.f,0.1f,-1.f,0.f,0.1f,0.f,0.f,0.f,0.f,0.f,0.f,0.f,0.f,0.2f,0.5f};
  const float hi[15] = {1.f,0.9f, 1.f,1.f,0.9f,1.f,5.f,0.5f,0.5f,1.f,1.f,1.f,1.f,1.f,2.f};
  float p[15];
#pragma unroll
  for (int i = 0; i < 15; ++i) p[i] = params[b * 15 + i] * (hi[i] - lo[i]) + lo[i];
  float fg = (float)mel * (1.f / 127.f);
  float tg = (float)t * (1.f / 255.f);
  float fd = fg - p[0];
  float sgn = (fd > 0.f) ? 1.f : ((fd < 0.f) ? -1.f : 0.f);
  float sigma = fmaxf(p[1] * (1.f + p[2] * sgn * 0.5f), 0.05f);
  float fr = fd / sigma;
  float fprof = expf(-0.5f * fr * fr);
  float td = fabsf(tg - p[3]);
  float tr = td / fmaxf(p[4], 0.05f);
  float tgs = expf(-0.5f * tr * tr);
  float tfl = (td < p[4]) ? 1.f : 0.f;
  float tprof = tgs * (1.f - p[5]) + tfl * p[5];
  float spec = fprof * tprof;
  if      (fg < 0.25f) spec *= p[9];
  else if (fg < 0.50f) spec *= p[10];
  else if (fg < 0.75f) spec *= p[11];
  else if (fg < 1.00f) spec *= p[12];
  float texture = p[7] * sinf(TWO_PI * p[6] * tg * 10.f) * sinf(TWO_PI * p[6] * fg * 5.f);
  spec *= (1.f + texture);
  spec *= (1.f + (noise[idx] - 0.5f) * p[8]);
  spec = ((spec - 0.5f) * p[14] + 0.5f) * p[13];
  spec = fminf(fmaxf(spec, 0.f), 1.f);
  specp[idx] = spec * spec * 100.f;   // spec_power
}

// ------------- linear resize 128->513 on freq axis + sqrt -> mag -----------
// mag layout: [b][t][KHALF]  (k >= 513 zero-padded)
__global__ void resize_mag_kernel(const float* __restrict__ specp, float* __restrict__ mag) {
  int idx = blockIdx.x * blockDim.x + threadIdx.x;
  if (idx >= MROWS * KHALF) return;
  int k = idx % KHALF, m = idx / KHALF;
  int t = m & 255, b = m >> 8;
  float v = 0.f;
  if (k < FREQ_BINS) {
    float src = ((float)k + 0.5f) * (128.0f / 513.0f) - 0.5f;
    src = fminf(fmaxf(src, 0.f), 127.f);
    int i0 = (int)src;
    int i1 = (i0 + 1 < 127) ? i0 + 1 : 127;
    float f = src - (float)i0;
    float a = specp[((size_t)(b * N_MELS + i0)) * N_FRAMES + t];
    float c = specp[((size_t)(b * N_MELS + i1)) * N_FRAMES + t];
    float lin = a + (c - a) * f;
    v = sqrtf(fmaxf(lin, 1e-12f));
  }
  mag[idx] = v;
}

// ----- initial A-matrix for iSTFT:  S = mag * exp(2*pi*i*phase_init) -------
__global__ void init_aist_kernel(const float* __restrict__ phase,
                                 const float* __restrict__ mag,
                                 _Float16* __restrict__ Aist) {
  int idx = blockIdx.x * blockDim.x + threadIdx.x;
  if (idx >= MROWS * KHALF) return;
  int k = idx % KHALF, m = idx / KHALF;
  int t = m & 255, b = m >> 8;
  size_t base = (size_t)m * KSPEC;
  if (k < FREQ_BINS) {
    float pv = phase[((size_t)(b * FREQ_BINS + k)) * N_FRAMES + t];
    float s, c;
    sincosf(TWO_PI * pv, &s, &c);
    float mg = mag[idx];
    Aist[base + k]         = (_Float16)(c * mg);
    Aist[base + KHALF + k] = (_Float16)(s * mg);
  } else {
    Aist[base + k]         = (_Float16)0.f;
    Aist[base + KHALF + k] = (_Float16)0.f;
  }
}

// ------------------------- overlap-add helpers -----------------------------
__device__ inline float oa_sample(const float* __restrict__ frames,
                                  const float* __restrict__ win, int b, int s) {
  int fh = s >> 8; if (fh > 255) fh = 255;
  int fl = fh - 3; if (fl < 0) fl = 0;
  float acc = 0.f, wsq = 0.f;
  for (int f = fl; f <= fh; ++f) {
    int n = s - (f << 8);
    if (n >= NFFT) continue;
    float w = win[n];
    acc += frames[(((size_t)(b * N_FRAMES + f)) << 10) + n];
    wsq += w * w;
  }
  return acc / fmaxf(wsq, 1e-11f);
}

// OLA + wsq-normalize into the interior of the reflect-padded buffer xp
__global__ void oa_kernel(const float* __restrict__ frames,
                          const float* __restrict__ win, float* __restrict__ xp) {
  int idx = blockIdx.x * blockDim.x + threadIdx.x;
  if (idx >= B_SZ * OUT_LEN) return;
  int b = idx / OUT_LEN, i = idx % OUT_LEN;
  int s = i + PADL;
  xp[(size_t)b * FULL_LEN + s] = oa_sample(frames, win, b, s);
}

// fill reflect pads of xp
__global__ void pad_kernel(float* __restrict__ xp) {
  int idx = blockIdx.x * blockDim.x + threadIdx.x;
  if (idx >= B_SZ * PADL) return;
  int j = idx & 511, b = idx >> 9;
  float* x = xp + (size_t)b * FULL_LEN;
  x[j] = x[1024 - j];                      // left reflect
  x[PADL + OUT_LEN + j] = x[65790 - j];    // right reflect
}

// frames for STFT GEMM (window folded into CmatT): f16 gather from xp
__global__ void build_frames_kernel(const float* __restrict__ xp, _Float16* __restrict__ fIn) {
  int idx = blockIdx.x * blockDim.x + threadIdx.x;
  if (idx >= MROWS * NFFT) return;
  int n = idx & 1023, m = idx >> 10;
  int f = m & 255, b = m >> 8;
  fIn[idx] = (_Float16)xp[(size_t)b * FULL_LEN + f * HOP + n];
}

// momentum + phase renormalize + build next iSTFT A-matrix (mag*angles, f16)
__global__ void update_kernel(const float* __restrict__ Rb, float* __restrict__ tp,
                              const float* __restrict__ mag, _Float16* __restrict__ Aist) {
  int idx = blockIdx.x * blockDim.x + threadIdx.x;
  if (idx >= MROWS * KHALF) return;
  int k = idx % KHALF, m = idx / KHALF;
  size_t base = (size_t)m * KSPEC;
  if (k < FREQ_BINS) {
    float re = Rb[base + k], im = Rb[base + KHALF + k];
    float nr = re - MOMC * tp[base + k];
    float ni = im - MOMC * tp[base + KHALF + k];
    float a = sqrtf(nr * nr + ni * ni) + 1e-16f;
    float s = mag[idx] / a;
    Aist[base + k]         = (_Float16)(nr * s);
    Aist[base + KHALF + k] = (_Float16)(ni * s);
    tp[base + k]           = re;
    tp[base + KHALF + k]   = im;
  } else {
    Aist[base + k]         = (_Float16)0.f;
    Aist[base + KHALF + k] = (_Float16)0.f;
  }
}

// final OLA -> d_out (+zero tail) + per-batch |max| via exact atomic max
__global__ void final_kernel(const float* __restrict__ frames, const float* __restrict__ win,
                             float* __restrict__ out, float* __restrict__ maxbuf) {
  int idx = blockIdx.x * blockDim.x + threadIdx.x;
  if (idx >= B_SZ * NUM_SAMPLES) return;
  int i = idx & (NUM_SAMPLES - 1), b = idx >> 16;
  if (i < OUT_LEN) {
    float v = oa_sample(frames, win, b, i + PADL);
    out[idx] = v;
    atomicMax((unsigned int*)(maxbuf + b), __float_as_uint(fabsf(v)));  // v>=0 bits ordered
  } else {
    out[idx] = 0.f;
  }
}

__global__ void scale_kernel(float* __restrict__ out, const float* __restrict__ maxbuf) {
  int idx = blockIdx.x * blockDim.x + threadIdx.x;
  if (idx >= B_SZ * NUM_SAMPLES) return;
  int b = idx >> 16;
  float mx = fmaxf(maxbuf[b], 1e-8f);
  out[idx] = out[idx] / mx * 0.9f;
}

// ---------------------------------------------------------------------------
static inline void launch_gemm(const _Float16* A, const _Float16* Bt, float* C,
                               int M, int Nvalid, int K, int lda, int ldb, int ldc,
                               hipStream_t stream) {
  int Mt = M / 64, Nt = cdiv(Nvalid, 64);
  int tiles = Mt * Nt;
  int blocks = cdiv(tiles * 32, 256);
  gemm_wmma<<<blocks, 256, 0, stream>>>(A, Bt, C, Nvalid, K, lda, ldb, ldc, Mt, Nt);
}

extern "C" void kernel_launch(void* const* d_in, const int* in_sizes, int n_in,
                              void* d_out, int out_size, void* d_ws, size_t ws_size,
                              hipStream_t stream) {
  (void)in_sizes; (void)n_in; (void)out_size;
  const float* params = (const float*)d_in[0];
  const float* noise  = (const float*)d_in[1];
  const float* phase  = (const float*)d_in[2];
  float* out = (float*)d_out;
  char* ws = (char*)d_ws;

  size_t off = 0;
  auto alloc = [&](size_t bytes) {
    size_t o = off;
    off += (bytes + 255) & ~(size_t)255;
    return o;
  };
  size_t o_mag  = alloc((size_t)MROWS * KHALF * 4);      // f32 [m][528]
  size_t o_aist = alloc((size_t)MROWS * KSPEC * 2);      // f16 [m][1056]
  size_t o_dmt  = alloc((size_t)NFFT  * KSPEC * 2);      // f16 [1024][1056]
  size_t o_cmt  = alloc((size_t)NSPEC_PAD * NFFT * 2);   // f16 [1088][1024]
  size_t o_fr   = alloc((size_t)MROWS * NFFT * 4);       // f32 [m][1024]
  size_t o_xp   = alloc((size_t)B_SZ * FULL_LEN * 4);    // f32 [b][66304]
  size_t o_fin  = alloc((size_t)MROWS * NFFT * 2);       // f16 [m][1024]
  size_t o_rb   = alloc((size_t)MROWS * KSPEC * 4);      // f32 [m][1056]
  size_t o_tp   = alloc((size_t)MROWS * KSPEC * 4);      // f32 [m][1056]
  size_t o_sp   = alloc((size_t)B_SZ * N_MELS * N_FRAMES * 4);
  size_t o_win  = alloc((size_t)NFFT * 4);
  size_t o_max  = alloc((size_t)B_SZ * 4);
  if (off > ws_size) return;   // workspace too small; nothing safe to do

  float*     mag   = (float*)(ws + o_mag);
  _Float16*  Aist  = (_Float16*)(ws + o_aist);
  _Float16*  DmT   = (_Float16*)(ws + o_dmt);
  _Float16*  CmT   = (_Float16*)(ws + o_cmt);
  float*     frames= (float*)(ws + o_fr);
  float*     xp    = (float*)(ws + o_xp);
  _Float16*  fIn   = (_Float16*)(ws + o_fin);
  float*     Rb    = (float*)(ws + o_rb);
  float*     tp    = (float*)(ws + o_tp);
  float*     specp = (float*)(ws + o_sp);
  float*     win   = (float*)(ws + o_win);
  float*     maxbf = (float*)(ws + o_max);

  const int T = 256;
  // ---- setup ----
  win_kernel<<<cdiv(NFFT, T), T, 0, stream>>>(win);
  build_dmatT<<<cdiv(NFFT * KSPEC, T), T, 0, stream>>>(win, DmT);
  build_cmatT<<<cdiv(NSPEC_PAD * NFFT, T), T, 0, stream>>>(win, CmT);
  spec_kernel<<<cdiv(B_SZ * N_MELS * N_FRAMES, T), T, 0, stream>>>(params, noise, specp);
  resize_mag_kernel<<<cdiv(MROWS * KHALF, T), T, 0, stream>>>(specp, mag);
  init_aist_kernel<<<cdiv(MROWS * KHALF, T), T, 0, stream>>>(phase, mag, Aist);
  hipMemsetAsync(tp, 0, (size_t)MROWS * KSPEC * 4, stream);
  hipMemsetAsync(maxbf, 0, (size_t)B_SZ * 4, stream);

  // ---- Griffin-Lim iterations ----
  for (int it = 0; it < N_ITER; ++it) {
    // iSTFT: frames = Aist [8192x1056] * D [1056x1024]
    launch_gemm(Aist, DmT, frames, MROWS, NFFT, KSPEC, KSPEC, KSPEC, NFFT, stream);
    oa_kernel<<<cdiv(B_SZ * OUT_LEN, T), T, 0, stream>>>(frames, win, xp);
    pad_kernel<<<cdiv(B_SZ * PADL, T), T, 0, stream>>>(xp);
    build_frames_kernel<<<cdiv(MROWS * NFFT, T), T, 0, stream>>>(xp, fIn);
    // STFT: Rb = fIn [8192x1024] * C [1024x1056]
    launch_gemm(fIn, CmT, Rb, MROWS, KSPEC, NFFT, NFFT, NFFT, KSPEC, stream);
    update_kernel<<<cdiv(MROWS * KHALF, T), T, 0, stream>>>(Rb, tp, mag, Aist);
  }

  // ---- final iSTFT + normalize ----
  launch_gemm(Aist, DmT, frames, MROWS, NFFT, KSPEC, KSPEC, KSPEC, NFFT, stream);
  final_kernel<<<cdiv(B_SZ * NUM_SAMPLES, T), T, 0, stream>>>(frames, win, out, maxbf);
  scale_kernel<<<cdiv(B_SZ * NUM_SAMPLES, T), T, 0, stream>>>(out, maxbf);
}
